// VectorPoolAggregationModule_43645457662574
// MI455X (gfx1250) — matrix operand order repaired and checked by reading, not compiled, not run on hardware
//
#include <hip/hip_runtime.h>
#include <hip/hip_fp16.h>

// ---------------------------------------------------------------------------
// VectorPoolAggregation for MI455X (gfx1250, wave32, WMMA).
//   prep:   channel-reduce feats (f32) + pack w_group/w_post into f16
//           WMMA B-fragment order (per-lane contiguous chunks).
//   interp: per-center 3-NN search over LDS-tiled support cloud, IDW interp,
//           writes f16 A-rows padded K 41->64 (zeros).
//   gconv:  per (g, mtile, ntile) wave: 2x v_wmma_f32_16x16x32_f16 (K=64),
//           fused BN1+ReLU, f16 output.
//   post:   per (mtile, ntile) wave: 27x chained WMMA over K=864,
//           fused BN2+ReLU, f32 output.
// Workspace: 6 164 480 bytes.
// ---------------------------------------------------------------------------

typedef __attribute__((ext_vector_type(16))) _Float16 v16h;
typedef __attribute__((ext_vector_type(8)))  _Float16 v8h;
typedef __attribute__((ext_vector_type(8)))  float    v8f;

#define B_     2
#define N_     2048
#define M_     512
#define G_     27
#define C_IN   128
#define C_RED  32
#define C_AGG  32
#define C_POST 128
#define CIG    41
#define KPAD   64
#define GC     (G_ * C_AGG)   // 864
#define BM     (B_ * M_)      // 1024
#define QR_    4.8f

// workspace offsets (bytes, 256-aligned)
#define WS_FEATS 0u            // 4096*32*4          = 524288
#define WS_VEC   524288u       // 1024*27*64*2       = 3538944
#define WS_WB    4063232u      // 27*2*2*32*16*2     = 110592
#define WS_WPB   4173824u      // 8*27*32*16*2       = 221184
#define WS_X     4395008u      // 1024*864*2         = 1769472  -> end 6164480

__device__ __forceinline__ v16h frag2(const _Float16* p0, const _Float16* p1) {
  v8h lo = *(const v8h*)p0;
  v8h hi = *(const v8h*)p1;
  return __builtin_shufflevector(lo, hi, 0,1,2,3,4,5,6,7,8,9,10,11,12,13,14,15);
}

// ---------------------------------------------------------------------------
// prep: id < 55296  -> wB  fragment pack  [g][ntile][step][lane][j]
//       id < 165888 -> wpB fragment pack  [ntile][s][lane][j]
//       rest        -> feats channel reduction (4096x32)
// ---------------------------------------------------------------------------
__global__ __launch_bounds__(256) void prep_kernel(
    const float* __restrict__ w_group, const float* __restrict__ w_post,
    const float* __restrict__ support_features,
    _Float16* __restrict__ wB, _Float16* __restrict__ wpB,
    float* __restrict__ feats) {
  const int NW1 = G_ * 2048;        // 55296
  const int NW2 = 8 * 27 * 512;     // 110592
  int id = blockIdx.x * 256 + threadIdx.x;
  if (id < NW1) {
    int j = id & 15, lane = (id >> 4) & 31;
    int step = (id >> 9) & 1, ntile = (id >> 10) & 1, g = id >> 11;
    int o = ntile * 16 + (lane & 15);
    int K = step * 32 + (lane >> 4) * 8 + (j < 8 ? j : j + 8);   // B elem j <-> row K
    float v = (K < CIG) ? w_group[(g * C_AGG + o) * CIG + K] : 0.f;
    wB[id] = (_Float16)v;
  } else if (id < NW1 + NW2) {
    int id2 = id - NW1;
    int j = id2 & 15, lane = (id2 >> 4) & 31;
    int s = (id2 >> 9) % 27, ntile = id2 / 13824;
    int o = ntile * 16 + (lane & 15);
    int K = s * 32 + (lane >> 4) * 8 + (j < 8 ? j : j + 8);      // K < 864 always
    wpB[id2] = (_Float16)w_post[o * GC + K];
  } else {
    int id3 = id - NW1 - NW2;                         // < 131072
    int c = id3 & 31, p = id3 >> 5;
    float s = 0.f;
#pragma unroll
    for (int q = 0; q < 4; ++q) s += support_features[p * C_IN + q * C_RED + c];
    feats[id3] = s;
  }
}

// ---------------------------------------------------------------------------
// interp: one thread per (b,m,g) center; support xyz tiled through LDS.
// ---------------------------------------------------------------------------
__global__ __launch_bounds__(256) void interp_kernel(
    const float* __restrict__ support_xyz, const float* __restrict__ new_xyz,
    const float* __restrict__ feats, _Float16* __restrict__ vec) {
  __shared__ float sx[256], sy[256], sz[256];
  const int tid = threadIdx.x;
  const int idx = blockIdx.x * 256 + tid;
  const int b = idx / (M_ * G_);          // uniform per block (13824 % 256 == 0)
  const int within = idx % (M_ * G_);
  const int m = within / G_;
  const int g = within % G_;
  const int bm = b * M_ + m;

  const float cell = 1.6f;                // 2R/3
  int gx = g / 9, gy = (g / 3) % 3, gz = g % 3;
  float cx = new_xyz[bm * 3 + 0] + ((float)gx + 0.5f) * cell - 2.4f;
  float cy = new_xyz[bm * 3 + 1] + ((float)gy + 0.5f) * cell - 2.4f;
  float cz = new_xyz[bm * 3 + 2] + ((float)gz + 0.5f) * cell - 2.4f;

  const float INF = __builtin_inff();
  const int base_pt = b * N_;
  float d0 = INF, d1 = INF, d2 = INF;
  int i0 = base_pt, i1 = base_pt, i2 = base_pt;   // fallback idx for empty slots

  for (int t = 0; t < N_ / 256; ++t) {
    int pg = base_pt + t * 256 + tid;
    sx[tid] = support_xyz[pg * 3 + 0];
    sy[tid] = support_xyz[pg * 3 + 1];
    sz[tid] = support_xyz[pg * 3 + 2];
    __syncthreads();
#pragma unroll 4
    for (int i = 0; i < 256; ++i) {
      float dx = cx - sx[i], dy = cy - sy[i], dz = cz - sz[i];
      float ch = fmaxf(fabsf(dx), fmaxf(fabsf(dy), fabsf(dz)));
      float dd = dx * dx + dy * dy + dz * dz;
      if ((ch <= QR_) && (dd < d2)) {          // strict < keeps lowest idx on ties
        int pi = base_pt + t * 256 + i;
        if (dd < d1) {
          d2 = d1; i2 = i1;
          if (dd < d0) { d1 = d0; i1 = i0; d0 = dd; i0 = pi; }
          else         { d1 = dd; i1 = pi; }
        } else { d2 = dd; i2 = pi; }
      }
    }
    __syncthreads();
  }

  bool empty = !(d0 < INF);
  float r0 = (d0 < INF) ? 1.f / (d0 + 1e-8f) : 0.f;  // IDW on *squared* dist (ref)
  float r1 = (d1 < INF) ? 1.f / (d1 + 1e-8f) : 0.f;
  float r2 = (d2 < INF) ? 1.f / (d2 + 1e-8f) : 0.f;
  float inv = 1.f / fmaxf(r0 + r1 + r2, 1e-8f);
  float w0 = r0 * inv, w1 = r1 * inv, w2 = r2 * inv;
  if (empty) { w0 = w1 = w2 = 0.f; }

  _Float16* out = vec + (size_t)(bm * G_ + g) * KPAD;
  const float* f0 = feats + (size_t)i0 * C_RED;
  const float* f1 = feats + (size_t)i1 * C_RED;
  const float* f2 = feats + (size_t)i2 * C_RED;
#pragma unroll
  for (int c = 0; c < C_RED; ++c)
    out[c] = (_Float16)(w0 * f0[c] + w1 * f1[c] + w2 * f2[c]);

  int ids[3] = {i0, i1, i2};
#pragma unroll
  for (int k = 0; k < 3; ++k) {
    float lx = cx - support_xyz[ids[k] * 3 + 0];
    float ly = cy - support_xyz[ids[k] * 3 + 1];
    float lz = cz - support_xyz[ids[k] * 3 + 2];
    out[C_RED + k * 3 + 0] = empty ? (_Float16)0.f : (_Float16)lx;
    out[C_RED + k * 3 + 1] = empty ? (_Float16)0.f : (_Float16)ly;
    out[C_RED + k * 3 + 2] = empty ? (_Float16)0.f : (_Float16)lz;
  }
#pragma unroll
  for (int k = CIG; k < KPAD; ++k) out[k] = (_Float16)0.f;   // zero K-pad
}

// ---------------------------------------------------------------------------
// gconv: wave tile = (g, mtile, ntile); 3456 waves; K=64 -> 2 WMMAs.
// ---------------------------------------------------------------------------
__global__ __launch_bounds__(256) void gconv_kernel(
    const _Float16* __restrict__ vec, const _Float16* __restrict__ wB,
    const float* __restrict__ g1, const float* __restrict__ b1,
    const float* __restrict__ mu1, const float* __restrict__ var1,
    _Float16* __restrict__ xout) {
  int lane = threadIdx.x & 31;
  int w = blockIdx.x * 8 + (threadIdx.x >> 5);
  int g = w >> 7, rem = w & 127;
  int mtile = rem >> 1, ntile = rem & 1;
  int half = lane >> 4, lm = lane & 15;

  int gc = g * C_AGG + ntile * 16 + lm;          // this lane's output channel
  float scale = g1[gc] * rsqrtf(var1[gc] + 1e-5f);
  float shift = b1[gc] - mu1[gc] * scale;

  v8f acc = {};
#pragma unroll
  for (int step = 0; step < 2; ++step) {
    int m = mtile * 16 + lm;
    const _Float16* a = vec + (size_t)(m * G_ + g) * KPAD + half * 8 + step * 32;
    v16h af = frag2(a, a + 16);                                  // A: ISA 16-bit layout
    const _Float16* bb = wB + (size_t)((((g * 2) + ntile) * 2 + step) * 32 + lane) * 16;
    v16h bf = frag2(bb, bb + 8);                                 // pre-swizzled B chunk
    acc = __builtin_amdgcn_wmma_f32_16x16x32_f16(false, af, false, bf,
                                                 (short)0, acc, false, false);
  }
#pragma unroll
  for (int r = 0; r < 8; ++r) {
    int m = mtile * 16 + r + half * 8;           // D layout: VGPR r -> row r / r+8
    float v = fmaxf(acc[r] * scale + shift, 0.f);
    xout[(size_t)m * GC + gc] = (_Float16)v;
  }
}

// ---------------------------------------------------------------------------
// post: wave tile = (mtile, ntile); 512 waves; K=864 -> 27 chained WMMAs.
// ---------------------------------------------------------------------------
__global__ __launch_bounds__(256) void post_kernel(
    const _Float16* __restrict__ x, const _Float16* __restrict__ wpB,
    const float* __restrict__ g2, const float* __restrict__ b2,
    const float* __restrict__ mu2, const float* __restrict__ var2,
    float* __restrict__ out) {
  int lane = threadIdx.x & 31;
  int w = blockIdx.x * 8 + (threadIdx.x >> 5);
  int mtile = w >> 3, ntile = w & 7;
  int half = lane >> 4, lm = lane & 15;

  int oc = ntile * 16 + lm;
  float scale = g2[oc] * rsqrtf(var2[oc] + 1e-5f);
  float shift = b2[oc] - mu2[oc] * scale;

  int m = mtile * 16 + lm;
  const _Float16* arow = x + (size_t)m * GC + half * 8;
  const _Float16* bcol = wpB + (size_t)(ntile * 27) * 512 + lane * 16;

  v8f acc = {};
#pragma unroll
  for (int s = 0; s < 27; ++s) {
    const _Float16* a = arow + s * 32;
    v16h af = frag2(a, a + 16);
    v16h bf = frag2(bcol + s * 512, bcol + s * 512 + 8);
    acc = __builtin_amdgcn_wmma_f32_16x16x32_f16(false, af, false, bf,
                                                 (short)0, acc, false, false);
  }
#pragma unroll
  for (int r = 0; r < 8; ++r) {
    int mm = mtile * 16 + r + half * 8;
    out[(size_t)mm * C_POST + oc] = fmaxf(acc[r] * scale + shift, 0.f);
  }
}

// ---------------------------------------------------------------------------
extern "C" void kernel_launch(void* const* d_in, const int* in_sizes, int n_in,
                              void* d_out, int out_size, void* d_ws, size_t ws_size,
                              hipStream_t stream) {
  const float* support_xyz      = (const float*)d_in[0];
  const float* support_features = (const float*)d_in[1];
  const float* new_xyz          = (const float*)d_in[2];
  const float* w_group          = (const float*)d_in[3];
  const float* bn1_gamma        = (const float*)d_in[4];
  const float* bn1_beta         = (const float*)d_in[5];
  const float* bn1_mean         = (const float*)d_in[6];
  const float* bn1_var          = (const float*)d_in[7];
  const float* w_post           = (const float*)d_in[8];
  const float* bn2_gamma        = (const float*)d_in[9];
  const float* bn2_beta         = (const float*)d_in[10];
  const float* bn2_mean         = (const float*)d_in[11];
  const float* bn2_var          = (const float*)d_in[12];
  (void)in_sizes; (void)n_in; (void)out_size; (void)ws_size;

  char* ws = (char*)d_ws;                       // needs 6 164 480 bytes
  float*     feats = (float*)(ws + WS_FEATS);
  _Float16*  vec   = (_Float16*)(ws + WS_VEC);
  _Float16*  wB    = (_Float16*)(ws + WS_WB);
  _Float16*  wpB   = (_Float16*)(ws + WS_WPB);
  _Float16*  xmid  = (_Float16*)(ws + WS_X);

  prep_kernel  <<<1160, 256, 0, stream>>>(w_group, w_post, support_features,
                                          wB, wpB, feats);
  interp_kernel<<<108,  256, 0, stream>>>(support_xyz, new_xyz, feats, vec);
  gconv_kernel <<<432,  256, 0, stream>>>(vec, wB, bn1_gamma, bn1_beta,
                                          bn1_mean, bn1_var, xmid);
  post_kernel  <<<64,   256, 0, stream>>>(xmid, wpB, bn2_gamma, bn2_beta,
                                          bn2_mean, bn2_var, (float*)d_out);
}
